// Generator_55868934586771
// MI455X (gfx1250) — matrix operand order repaired and checked by reading, not compiled, not run on hardware
//
#include <hip/hip_runtime.h>
#include <hip/hip_bf16.h>

typedef __bf16 bf16_t;
typedef __attribute__((ext_vector_type(16))) __bf16 v16bf;
typedef __attribute__((ext_vector_type(8)))  __bf16 v8bf;
typedef __attribute__((ext_vector_type(8)))  float  v8f;

#define BM 128
#define BN 128
#define BK 32
#define LDSS 40   // 32 + 8 halfs pad -> 80B row stride, conflict-free b128 reads

// ---------------------------------------------------------------------------
// General WMMA GEMM: C[n,m] = act( alpha * sum_k A[n,k]*W[m,k] + bias[m] )
// A: [Nrows x K] bf16 row-major, W: [M x K] bf16 row-major (i.e. B^T).
// Block tile 128x128, 8 waves in 4x2; each wave computes 32x64 (8 WMMAs/K-step).
// Optional batch (blockIdx.z) with element strides sA/sW/sC.
// Output to Cf (f32) or Cb (bf16); optional transposed store (col-major).
// ---------------------------------------------------------------------------
__global__ __launch_bounds__(256)
void gemm_bf16_wmma(const bf16_t* __restrict__ A, const bf16_t* __restrict__ W,
                    const float* __restrict__ bias,
                    float* __restrict__ Cf, bf16_t* __restrict__ Cb,
                    int M, int K,
                    long long sA, long long sW, long long sC, int ldc,
                    const float* __restrict__ scale0,
                    const float* __restrict__ scale1,
                    int sidx, int act, int zero_diag, int tstore)
{
    __shared__ bf16_t As[BM * LDSS];
    __shared__ bf16_t Ws[BN * LDSS];

    const int tid  = threadIdx.x;
    const int wave = tid >> 5;
    const int lane = tid & 31;
    const int wr   = wave >> 1;      // 0..3 : row group (32 rows each)
    const int wc   = wave & 1;       // 0..1 : col group (64 cols each)

    const long long batch = blockIdx.z;
    A += batch * sA;
    W += batch * sW;

    const int row0 = blockIdx.y * BM;
    const int col0 = blockIdx.x * BN;

    v8f acc[8] = {};                 // [ar*4 + cg]

    // cooperative staging indices: 256 threads, 16B (8 halfs) per load
    const int ldr = tid >> 2;        // 0..63
    const int ldk = (tid & 3) * 8;   // 0,8,16,24

    v8bf zero8;
    #pragma unroll
    for (int i = 0; i < 8; ++i) zero8[i] = (__bf16)0.0f;

    const int  wrow0 = col0 + ldr;
    const int  wrow1 = col0 + ldr + 64;
    const bool wok0  = (wrow0 < M);
    const bool wok1  = (wrow1 < M);

    const int nl  = lane & 15;
    const int kbA = (lane < 16) ? 0 : 8;    // A: K = {kb..kb+7, kb+16..kb+23}
    const int kbB = (lane < 16) ? 0 : 8;    // B: K = {kb..kb+7, kb+16..kb+23}

    for (int k0 = 0; k0 < K; k0 += BK) {
        v8bf a0 = *(const v8bf*)(A + (long long)(row0 + ldr)      * K + k0 + ldk);
        v8bf a1 = *(const v8bf*)(A + (long long)(row0 + ldr + 64) * K + k0 + ldk);
        v8bf w0 = wok0 ? *(const v8bf*)(W + (long long)wrow0 * K + k0 + ldk) : zero8;
        v8bf w1 = wok1 ? *(const v8bf*)(W + (long long)wrow1 * K + k0 + ldk) : zero8;

        __syncthreads();   // previous iteration's LDS reads done
        *(v8bf*)&As[ldr * LDSS + ldk]        = a0;
        *(v8bf*)&As[(ldr + 64) * LDSS + ldk] = a1;
        *(v8bf*)&Ws[ldr * LDSS + ldk]        = w0;
        *(v8bf*)&Ws[(ldr + 64) * LDSS + ldk] = w1;
        __syncthreads();

        // load all fragments first so one dscnt-wait covers the WMMA chain
        v16bf af[2];
        #pragma unroll
        for (int ar = 0; ar < 2; ++ar) {
            const bf16_t* ap = &As[(wr * 32 + ar * 16 + nl) * LDSS + kbA];
            v8bf alo = *(const v8bf*)ap;
            v8bf ahi = *(const v8bf*)(ap + 16);
            af[ar] = __builtin_shufflevector(alo, ahi,
                       0,1,2,3,4,5,6,7,8,9,10,11,12,13,14,15);
        }
        v16bf bf[4];
        #pragma unroll
        for (int cg = 0; cg < 4; ++cg) {
            const bf16_t* bp = &Ws[(wc * 64 + cg * 16 + nl) * LDSS + kbB];
            v8bf blo = *(const v8bf*)bp;
            v8bf bhi = *(const v8bf*)(bp + 16);
            bf[cg] = __builtin_shufflevector(blo, bhi,
                       0,1,2,3,4,5,6,7,8,9,10,11,12,13,14,15);
        }
        #pragma unroll
        for (int ar = 0; ar < 2; ++ar)
            #pragma unroll
            for (int cg = 0; cg < 4; ++cg)
                acc[ar * 4 + cg] = __builtin_amdgcn_wmma_f32_16x16x32_bf16(
                    false, af[ar], false, bf[cg], (short)0, acc[ar * 4 + cg],
                    false, false);
    }

    float scl = 1.0f;
    if (scale0) scl *= scale0[sidx];
    if (scale1) scl *= scale1[sidx];
    const int moff = (lane < 16) ? 0 : 8;

    #pragma unroll
    for (int ar = 0; ar < 2; ++ar) {
        #pragma unroll
        for (int cg = 0; cg < 4; ++cg) {
            const int col = col0 + wc * 64 + cg * 16 + nl;
            if (col >= M) continue;
            const float bv = bias ? bias[col] : 0.0f;
            #pragma unroll
            for (int r = 0; r < 8; ++r) {
                const int row = row0 + wr * 32 + ar * 16 + moff + r;
                float v = acc[ar * 4 + cg][r] * scl + bv;
                if (act == 1)      v = fmaxf(v, 0.0f);
                else if (act == 2) v = 1.0f / (1.0f + __expf(-v));
                if (zero_diag && row == col) v = 0.0f;
                const long long idx = tstore ? ((long long)col * ldc + row)
                                             : ((long long)row * ldc + col);
                if (Cf) Cf[batch * sC + idx] = v;
                else    Cb[batch * sC + idx] = (bf16_t)v;
            }
        }
    }
}

// ---------------------------------------------------------------------------
// Elementwise / reduction helpers
// ---------------------------------------------------------------------------
__global__ void cvt_f32_bf16(const float* __restrict__ in, bf16_t* __restrict__ out,
                             long long n)
{
    for (long long i = (long long)blockIdx.x * blockDim.x + threadIdx.x; i < n;
         i += (long long)gridDim.x * blockDim.x)
        out[i] = (bf16_t)in[i];
}

__global__ void stats_init(float* s, float* q, int M)
{
    int i = blockIdx.x * blockDim.x + threadIdx.x;
    if (i < M) { s[i] = 0.0f; q[i] = 0.0f; }
}

__global__ void colstats(const float* __restrict__ Y, float* s, float* q,
                         int M, int rows_per_blk)
{
    int col = blockIdx.x * blockDim.x + threadIdx.x;
    if (col >= M) return;
    long long r0 = (long long)blockIdx.y * rows_per_blk;
    float sv = 0.0f, qv = 0.0f;
    for (int r = 0; r < rows_per_blk; ++r) {
        float v = Y[(r0 + r) * M + col];
        sv += v; qv += v * v;
    }
    atomicAdd(&s[col], sv);
    atomicAdd(&q[col], qv);
}

__global__ void bn_finalize(const float* s, const float* q,
                            const float* g, const float* b,
                            float* scale, float* shift, int M, float invN)
{
    int col = blockIdx.x * blockDim.x + threadIdx.x;
    if (col >= M) return;
    float mean = s[col] * invN;
    float var  = q[col] * invN - mean * mean;
    float rstd = rsqrtf(var + 1e-5f);
    float sc   = g[col] * rstd;
    scale[col] = sc;
    shift[col] = b[col] - mean * sc;
}

__global__ void bn_relu_bf16(const float* __restrict__ Y,
                             const float* __restrict__ scale,
                             const float* __restrict__ shift,
                             bf16_t* __restrict__ out, long long total, int M)
{
    for (long long i = (long long)blockIdx.x * blockDim.x + threadIdx.x; i < total;
         i += (long long)gridDim.x * blockDim.x) {
        int col = (int)(i % M);
        float v = Y[i] * scale[col] + shift[col];
        out[i] = (bf16_t)fmaxf(v, 0.0f);
    }
}

__global__ void h_update(const float* __restrict__ attn, bf16_t* __restrict__ hb,
                         const float* __restrict__ wr, int widx, float invE,
                         long long n)
{
    float c = wr[widx] * invE;
    for (long long i = (long long)blockIdx.x * blockDim.x + threadIdx.x; i < n;
         i += (long long)gridDim.x * blockDim.x)
        hb[i] = (bf16_t)(c * attn[i] + (float)hb[i]);
}

// ---------------------------------------------------------------------------
// Host-side orchestration
// ---------------------------------------------------------------------------
static void launch_gemm(hipStream_t stream,
                        const bf16_t* A, const bf16_t* W, const float* bias,
                        float* Cf, bf16_t* Cb,
                        int Nrows, int M, int K, int batches,
                        long long sA, long long sW, long long sC, int ldc,
                        const float* s0, const float* s1, int sidx,
                        int act, int zdiag, int tstore)
{
    dim3 grid((M + BN - 1) / BN, Nrows / BM, batches);
    gemm_bf16_wmma<<<grid, 256, 0, stream>>>(A, W, bias, Cf, Cb, M, K,
                                             sA, sW, sC, ldc, s0, s1, sidx,
                                             act, zdiag, tstore);
}

extern "C" void kernel_launch(void* const* d_in, const int* in_sizes, int n_in,
                              void* d_out, int out_size, void* d_ws, size_t ws_size,
                              hipStream_t stream)
{
    (void)in_sizes; (void)n_in; (void)out_size; (void)ws_size;
    const int B = 32, E = 512, F = 256, H2 = 512, H4 = 1024, C = 32;
    const long long N = (long long)B * E;           // 16384 rows

    const float* x      = (const float*)d_in[0];
    const float* enc_w1 = (const float*)d_in[1];
    const float* enc_b1 = (const float*)d_in[2];
    const float* bn1_g  = (const float*)d_in[3];
    const float* bn1_b  = (const float*)d_in[4];
    const float* enc_w2 = (const float*)d_in[5];
    const float* enc_b2 = (const float*)d_in[6];
    const float* bn2_g  = (const float*)d_in[7];
    const float* bn2_b  = (const float*)d_in[8];
    const float* enc_w3 = (const float*)d_in[9];
    const float* enc_b3 = (const float*)d_in[10];
    const float* rel_wu = (const float*)d_in[11];
    const float* rel_bu = (const float*)d_in[12];
    const float* rel_psi= (const float*)d_in[13];
    const float* rel_phi= (const float*)d_in[14];
    const float* rel_wr = (const float*)d_in[15];
    const float* dec_w1 = (const float*)d_in[16];
    const float* dec_b1 = (const float*)d_in[17];
    const float* dbn_g  = (const float*)d_in[18];
    const float* dbn_b  = (const float*)d_in[19];
    const float* dec_w2 = (const float*)d_in[20];
    const float* dec_b2 = (const float*)d_in[21];
    const float* br1_w  = (const float*)d_in[22];
    const float* br1_b  = (const float*)d_in[23];
    const float* br2_w  = (const float*)d_in[24];
    const float* br2_b  = (const float*)d_in[25];

    // bump allocator over workspace
    char*  ws  = (char*)d_ws;
    size_t off = 0;
    auto alloc = [&](size_t bytes) -> char* {
        char* p = ws + off;
        off = (off + bytes + 255) & ~(size_t)255;
        return p;
    };

    bf16_t* w1b  = (bf16_t*)alloc((size_t)H2 * F * 2);
    bf16_t* w2b  = (bf16_t*)alloc((size_t)H4 * H2 * 2);
    bf16_t* w3b  = (bf16_t*)alloc((size_t)H4 * H4 * 2);
    bf16_t* wub  = (bf16_t*)alloc((size_t)4 * H4 * H4 * 2);
    bf16_t* dw1b = (bf16_t*)alloc((size_t)H2 * H4 * 2);
    bf16_t* dw2b = (bf16_t*)alloc((size_t)F * H2 * 2);
    bf16_t* bw1b = (bf16_t*)alloc((size_t)C * F * 2);
    bf16_t* bw2b = (bf16_t*)alloc((size_t)(F - C) * F * 2);
    float*  ssum = (float*)alloc(H4 * 4);
    float*  ssq  = (float*)alloc(H4 * 4);
    float*  bscale = (float*)alloc(H4 * 4);
    float*  bshift = (float*)alloc(H4 * 4);
    float*  Y    = (float*)alloc((size_t)N * H4 * 4);    // f32 GEMM out / attn
    bf16_t* buf1 = (bf16_t*)alloc((size_t)N * H4 * 2);   // A2b -> U^T -> D2b
    bf16_t* buf2 = (bf16_t*)alloc((size_t)N * H2 * 2);   // A1b -> S  -> D1b
    bf16_t* hb   = (bf16_t*)alloc((size_t)N * H4 * 2);   // Xb -> h (bf16)
    bf16_t* xb   = hb;                                   // Xb dead before h written

    auto cvt = [&](const float* src, bf16_t* dst, long long n) {
        int blocks = (int)((n + 255) / 256); if (blocks > 4096) blocks = 4096;
        cvt_f32_bf16<<<blocks, 256, 0, stream>>>(src, dst, n);
    };
    auto bnstats = [&](const float* Yp, const float* g, const float* b, int M) {
        stats_init<<<(M + 255) / 256, 256, 0, stream>>>(ssum, ssq, M);
        colstats<<<dim3((M + 255) / 256, 64), 256, 0, stream>>>(
            Yp, ssum, ssq, M, (int)(N / 64));
        bn_finalize<<<(M + 255) / 256, 256, 0, stream>>>(
            ssum, ssq, g, b, bscale, bshift, M, 1.0f / (float)N);
    };
    auto bnrelu = [&](const float* Yp, bf16_t* dst, int M) {
        bn_relu_bf16<<<4096, 256, 0, stream>>>(Yp, bscale, bshift, dst,
                                               N * (long long)M, M);
    };

    // ---- weight & input conversion to bf16 ----
    cvt(enc_w1, w1b,  (long long)H2 * F);
    cvt(enc_w2, w2b,  (long long)H4 * H2);
    cvt(enc_w3, w3b,  (long long)H4 * H4);
    cvt(rel_wu, wub,  (long long)4 * H4 * H4);
    cvt(dec_w1, dw1b, (long long)H2 * H4);
    cvt(dec_w2, dw2b, (long long)F * H2);
    cvt(br1_w,  bw1b, (long long)C * F);
    cvt(br2_w,  bw2b, (long long)(F - C) * F);
    cvt(x, xb, N * F);

    // ---- encoder ----
    launch_gemm(stream, xb, w1b, enc_b1, Y, nullptr, (int)N, H2, F, 1,
                0, 0, 0, H2, nullptr, nullptr, 0, 0, 0, 0);
    bnstats(Y, bn1_g, bn1_b, H2);
    bnrelu(Y, buf2, H2);                                         // A1b

    launch_gemm(stream, buf2, w2b, enc_b2, Y, nullptr, (int)N, H4, H2, 1,
                0, 0, 0, H4, nullptr, nullptr, 0, 0, 0, 0);
    bnstats(Y, bn2_g, bn2_b, H4);
    bnrelu(Y, buf1, H4);                                         // A2b

    // enc3 + sigmoid fused, bf16 out -> h (overwrites xb, which is dead)
    launch_gemm(stream, buf1, w3b, enc_b3, nullptr, hb, (int)N, H4, H4, 1,
                0, 0, 0, H4, nullptr, nullptr, 0, 2, 0, 0);

    // ---- relation layers (batched over B=32) ----
    const long long sH = (long long)E * H4;   // 512*1024
    const long long sS = (long long)E * E;    // 512*512
    for (int k = 0; k < 4; ++k) {
        const bf16_t* wu_k = wub + (long long)k * H4 * H4;
        // U^T[b] = (relu(h[b] @ Wu^T + bu))^T stored [H4 x E], transposed store
        launch_gemm(stream, hb, wu_k, rel_bu + (long long)k * H4,
                    nullptr, buf1, E, H4, H4, B,
                    sH, 0, sH, E, nullptr, nullptr, 0, 1, 0, 1);
        // S[b] = psi*phi * h[b] @ h[b]^T, diag zeroed, bf16
        launch_gemm(stream, hb, hb, nullptr, nullptr, buf2, E, E, H4, B,
                    sH, sH, sS, E, rel_psi, rel_phi, k, 0, 1, 0);
        // attn[b] = S[b] @ (U^T[b])^T  -> f32 in Y
        launch_gemm(stream, buf2, buf1, nullptr, Y, nullptr, E, H4, E, B,
                    sS, sH, sH, H4, nullptr, nullptr, 0, 0, 0, 0);
        // h = wr/E * attn + h
        h_update<<<4096, 256, 0, stream>>>(Y, hb, rel_wr, k, 1.0f / (float)E,
                                           N * (long long)H4);
    }

    // ---- decoder ----
    launch_gemm(stream, hb, dw1b, dec_b1, Y, nullptr, (int)N, H2, H4, 1,
                0, 0, 0, H2, nullptr, nullptr, 0, 0, 0, 0);
    bnstats(Y, dbn_g, dbn_b, H2);
    bnrelu(Y, buf2, H2);                                         // D1b

    launch_gemm(stream, buf2, dw2b, dec_b2, nullptr, buf1, (int)N, F, H2, 1,
                0, 0, 0, F, nullptr, nullptr, 0, 1, 0, 0);       // D2b (relu)

    // ---- output branches, concatenated along feature dim ----
    float* out = (float*)d_out;
    launch_gemm(stream, buf1, bw1b, br1_b, out, nullptr, (int)N, C, F, 1,
                0, 0, 0, F, nullptr, nullptr, 0, 0, 0, 0);
    launch_gemm(stream, buf1, bw2b, br2_b, out + C, nullptr, (int)N, F - C, F, 1,
                0, 0, 0, F, nullptr, nullptr, 0, 0, 0, 0);
}
